// EigenExpert_87540023427783
// MI455X (gfx1250) — matrix operand order repaired and checked by reading, not compiled, not run on hardware
//
#include <hip/hip_runtime.h>
#include <hip/hip_bf16.h>

// ---- CDNA5 WMMA vector types -------------------------------------------------
typedef __attribute__((ext_vector_type(16))) __bf16        bf16x16;
typedef __attribute__((ext_vector_type(8)))  float         v8f;
typedef __attribute__((ext_vector_type(8)))  unsigned int  u32x8;

__device__ __forceinline__ unsigned short bf16_bits(float f) {
    __bf16 b = (__bf16)f;
    return __builtin_bit_cast(unsigned short, b);
}

// A-operand K index for 16-bit 16x32 tiles (ISA 7.12.2):
// lanes 0-15 : VGPR v<4 -> K=2v,2v+1 ; v>=4 -> K=16+2(v-4)..
// lanes 16-31: same +8 within each half.
__device__ __forceinline__ int a_kindex(int hi, int kt, int v, int j) {
    int kl = (v < 4) ? (hi * 8 + 2 * v + j) : (16 + hi * 8 + 2 * (v - 4) + j);
    return kt * 32 + kl;
}

__global__ void eigen_expert_init(float* __restrict__ out,
                                  const float* __restrict__ b3) {
    out[threadIdx.x] = b3[0];   // mean(y) = mean(h2@W3) + b3
}

__global__ void __launch_bounds__(256)
eigen_expert_mlp(const float* __restrict__ U,    // [M] flat (N*64)
                 const float* __restrict__ W1,   // [128]
                 const float* __restrict__ b1,   // [128]
                 const float* __restrict__ W2,   // [128,128] row-major (h,n)
                 const float* __restrict__ b2,   // [128]
                 const float* __restrict__ W3,   // [128]
                 float* __restrict__ out,        // [64]
                 int nTiles, float invN) {
    // W2 staged in LDS, pre-swizzled into WMMA B-operand layout:
    // [kt][nt][lane][elem] ; lane = hi*16 + (n&15) ; elem i -> K = (i<8) ? hi*8+i : 16+hi*8+(i-8)
    __shared__ __align__(32) __bf16 w2s[4][8][32][16];   // 32 KB
    __shared__ float acc_s[64];

    const int tid  = threadIdx.x;
    const int lane = tid & 31;
    const int hi   = lane >> 4;       // half-wave select
    const int lm   = lane & 15;

    // ---- stage W2 (fp32 -> bf16, B layout) ---------------------------------
    for (int idx = tid; idx < 128 * 128; idx += blockDim.x) {
        int h = idx >> 7, n = idx & 127;
        int kt = h >> 5, kl = h & 31;
        w2s[kt][n >> 4][((kl >> 3) & 1) * 16 + (n & 15)][(kl & 7) | ((kl >> 4) << 3)] = (__bf16)W2[idx];
    }
    if (tid < 64) acc_s[tid] = 0.0f;

    // ---- per-lane packed W1/b1 pairs matching the A-operand element order --
    unsigned int w1pk[4][8], b1pk[4][8];
#pragma unroll
    for (int kt = 0; kt < 4; ++kt)
#pragma unroll
        for (int v = 0; v < 8; ++v) {
            int k0 = a_kindex(hi, kt, v, 0);
            int k1 = a_kindex(hi, kt, v, 1);
            w1pk[kt][v] = ((unsigned)bf16_bits(W1[k1]) << 16) | bf16_bits(W1[k0]);
            b1pk[kt][v] = ((unsigned)bf16_bits(b1[k1]) << 16) | bf16_bits(b1[k0]);
        }
    // per-lane layer-3 constants (C/D layout: column n = nt*16 + (lane&15))
    float b2l[8], w3l[8];
#pragma unroll
    for (int nt = 0; nt < 8; ++nt) {
        b2l[nt] = b2[nt * 16 + lm];
        w3l[nt] = W3[nt * 16 + lm];
    }
    __syncthreads();

    const int wave   = (blockIdx.x * blockDim.x + tid) >> 5;
    const int nWaves = (gridDim.x * blockDim.x) >> 5;  // multiple of 4!
    const unsigned int zero2 = 0u;

    // Because nWaves % 4 == 0, every tile this wave visits has the same
    // output-column group g = tile & 3  (e_base = 16*g).  So we can keep the
    // final y accumulation entirely in registers across the whole loop.
    float ysacc[8] = {0.f, 0.f, 0.f, 0.f, 0.f, 0.f, 0.f, 0.f};

    for (int t = wave; t < nTiles; t += nWaves) {
        const int rowBase = t << 4;
        float u = U[rowBase + lm];                 // row m = rowBase + (lane&15)
        int tn = t + nWaves;                        // prefetch next U tile
        if (tn < nTiles) __builtin_prefetch(&U[(tn << 4) + lm], 0, 1);

        // ---- layer 1: h1 = relu(u*W1+b1), built directly in A layout -------
        unsigned short us = bf16_bits(u);
        unsigned int upk = ((unsigned)us << 16) | us;
        bf16x16 A[4];
#pragma unroll
        for (int kt = 0; kt < 4; ++kt) {
            u32x8 ap;
#pragma unroll
            for (int v = 0; v < 8; ++v) {
                unsigned int h;
                asm("v_pk_fma_bf16 %0, %1, %2, %3"
                    : "=v"(h) : "v"(upk), "v"(w1pk[kt][v]), "v"(b1pk[kt][v]));
                asm("v_pk_max_num_bf16 %0, %1, %2"
                    : "=v"(h) : "v"(h), "v"(zero2));
                ap[v] = h;
            }
            A[kt] = __builtin_bit_cast(bf16x16, ap);
        }

        // ---- layer 2 (WMMA bf16, K=128) + fused layer-3 epilogue -----------
#pragma unroll
        for (int nt = 0; nt < 8; ++nt) {
            v8f c = {};
#pragma unroll
            for (int kt = 0; kt < 4; ++kt) {
                bf16x16 B = *(const bf16x16*)&w2s[kt][nt][lane][0];
                c = __builtin_amdgcn_wmma_f32_16x16x32_bf16(
                        false, A[kt], false, B, (short)0, c, false, false);
            }
#pragma unroll
            for (int r = 0; r < 8; ++r) {                 // row = r + 8*hi
                float h2 = fmaxf(c[r] + b2l[nt], 0.0f);
                ysacc[r] = fmaf(h2, w3l[nt], ysacc[r]);   // accumulated over ALL tiles
            }
        }
    }

    // ---- once per wave: reduce over the 128 columns, fold into LDS ---------
    const int eBase = (wave & 3) << 4;       // same for every tile of this wave
#pragma unroll
    for (int r = 0; r < 8; ++r) {
        float s = ysacc[r];
        s += __shfl_xor(s, 1, 16);
        s += __shfl_xor(s, 2, 16);
        s += __shfl_xor(s, 4, 16);
        s += __shfl_xor(s, 8, 16);
        if (lm == r) atomicAdd(&acc_s[eBase + r + (hi << 3)], s);
    }

    __syncthreads();
    if (tid < 64) atomicAdd(&out[tid], acc_s[tid] * invN);
}

extern "C" void kernel_launch(void* const* d_in, const int* in_sizes, int n_in,
                              void* d_out, int out_size, void* d_ws, size_t ws_size,
                              hipStream_t stream) {
    const float* U  = (const float*)d_in[0];
    const float* W1 = (const float*)d_in[1];
    const float* b1 = (const float*)d_in[2];
    const float* W2 = (const float*)d_in[3];
    const float* b2 = (const float*)d_in[4];
    const float* W3 = (const float*)d_in[5];
    const float* b3 = (const float*)d_in[6];
    float* out = (float*)d_out;

    const int M = in_sizes[0];          // N * 64
    const int nTiles = M >> 4;          // 16 rows per tile
    const float invN = 64.0f / (float)M;

    eigen_expert_init<<<1, 64, 0, stream>>>(out, b3);
    // 1024 blocks x 8 waves = 8192 waves (multiple of 4 -> fixed e-group/wave)
    eigen_expert_mlp<<<1024, 256, 0, stream>>>(U, W1, b1, W2, b2, W3, out,
                                               nTiles, invN);
}
